// JOGPSSM_66898410603095
// MI455X (gfx1250) — compile-verified, matrix operand order checked by reading, not run on hardware
//
#include <hip/hip_runtime.h>
#include <hip/hip_bf16.h>

typedef __attribute__((ext_vector_type(16))) _Float16 v16h;
typedef __attribute__((ext_vector_type(8)))  _Float16 v8h;
typedef __attribute__((ext_vector_type(8)))  float    v8f;
typedef __attribute__((ext_vector_type(4))) unsigned int u32x4;
typedef __attribute__((ext_vector_type(8))) int i32x8;
typedef __attribute__((ext_vector_type(4))) int i32x4;

#define TLEN 512
// group1 dword0 for f16 (data_size code 1 at bits[17:16])
#define TDM_F16 0x00010000u
// + pad_enable(bit20) + pad_interval=128DW(code6 at [24:22]) + pad_amount=16DW(code15 at [31:25])
#define TDM_F16_PAD288 (0x00010000u | (1u<<20) | (6u<<22) | (15u<<25))

__device__ __forceinline__ float sigmf_(float x){ return 1.f/(1.f+__expf(-x)); }
__device__ __forceinline__ float softplusf_(float x){ return (x > 20.f) ? x : __logf(1.f + __expf(x)); }

__device__ __forceinline__ v8f wmma16(v16h a, v16h b, v8f c) {
  return __builtin_amdgcn_wmma_f32_16x16x32_f16(false, a, false, b, (short)0, c, false, false);
}

// Row-major f16 matrix fragment load matching the CDNA5 16-bit A/B layout:
// lane<16 -> row (lane&15), K halves {kb..kb+7, kb+16..kb+23}; lane>=16 -> +8.
__device__ __forceinline__ v16h load_frag(const _Float16* p, size_t ld, size_t row, int kb, int lane) {
  const int sel = (lane & 16) ? 8 : 0;
  const _Float16* q = p + row*ld + (size_t)(kb + sel);
  v8h lo = *(const v8h*)q;
  v8h hi = *(const v8h*)(q + 16);
  v16h r;
#pragma unroll
  for (int i = 0; i < 8; ++i) { r[i] = lo[i]; r[8+i] = hi[i]; }
  return r;
}

// LDS byte offset of a (dynamic) shared-memory pointer via addrspacecast.
__device__ __forceinline__ unsigned lds_off(const void* p) {
  return (unsigned)(unsigned long long)(__attribute__((address_space(3))) const char*)p;
}

// Tensor Data Mover: 2-D tile (width x rows, f16) global -> LDS.
// D# groups per CDNA5 ISA ch.8; dims/strides in data_size units.
// This toolchain exposes the 6-arg builtin: (g0, g1, g2, g3, g4, cpol).
__device__ __forceinline__ void tdm_load_2d(unsigned ldsByteOff, const void* gptr,
                                            unsigned width, unsigned rows,
                                            unsigned long long rowStride,
                                            unsigned flags) {
  unsigned long long ga = (unsigned long long)(size_t)gptr;
  u32x4 g0;
  g0[0] = 1u;                                                 // count=1 (valid user D#)
  g0[1] = ldsByteOff;                                         // lds_addr
  g0[2] = (unsigned)ga;                                       // global_addr[31:0]
  g0[3] = (unsigned)((ga >> 32) & 0x01ffffffu) | (2u << 30);  // global_addr[56:32] | type=2
  i32x8 g1;
  g1[0] = (int)flags;                                         // wg_mask=0 | data_size | pad cfg
  g1[1] = (int)((width & 0xffffu) << 16);                     // tensor_dim0[15:0] @ [63:48]
  g1[2] = (int)(((width >> 16) & 0xffffu) | ((rows & 0xffffu) << 16)); // td0 hi | tensor_dim1 lo
  g1[3] = (int)(((rows >> 16) & 0xffffu) | ((width & 0xffffu) << 16)); // td1 hi | tile_dim0
  g1[4] = (int)(rows & 0xffffu);                              // tile_dim1 (tile_dim2=0)
  g1[5] = (int)(unsigned)(rowStride & 0xffffffffu);           // tensor_dim0_stride[31:0]
  g1[6] = (int)(unsigned)((rowStride >> 32) & 0xffffu);       // tensor_dim0_stride[47:32]
  g1[7] = 0;
  i32x4 gz = {0, 0, 0, 0};
  i32x8 gz8 = {0, 0, 0, 0, 0, 0, 0, 0};
  __builtin_amdgcn_tensor_load_to_lds(g0, g1, gz, gz, gz8, 0);
}

__device__ __forceinline__ float block_reduce_sum(float v, float* sred) {
  const int tid = threadIdx.x;
  sred[tid] = v;
  __syncthreads();
  for (int s = 128; s > 0; s >>= 1) {
    if (tid < s) sred[tid] += sred[tid + s];
    __syncthreads();
  }
  float r = sred[0];
  __syncthreads();
  return r;
}

// ---------------- small utility kernels ----------------
__global__ void k_cast_f16(const float* __restrict__ src, _Float16* __restrict__ dst, int n) {
  int i = blockIdx.x*256 + threadIdx.x;
  if (i < n) dst[i] = (_Float16)src[i];
}

__global__ void k_pack_w1(const float* __restrict__ W1, _Float16* __restrict__ out) {
  int i = blockIdx.x*256 + threadIdx.x;            // 128*288
  if (i < 128*288) {
    int n = i/288, k = i - n*288;
    out[i] = (k < 264) ? (_Float16)W1[n*264+k] : (_Float16)0.f;
  }
}

__global__ void k_pack_wmv(const float* __restrict__ Wm, const float* __restrict__ Wv,
                           _Float16* __restrict__ out) {
  int i = blockIdx.x*256 + threadIdx.x;            // 16*128
  if (i < 16*128) {
    int n = i >> 7, k = i & 127;
    out[i] = (_Float16)((n < 8) ? Wm[n*128+k] : Wv[(n-8)*128+k]);
  }
}

// ---------------- persistent batch-sliced LSTM ----------------
// One workgroup owns 16 batch rows for all T steps. Wh TDM-staged into LDS.
// xp path: per-step activation tile double-buffered via TDM one step ahead.
__global__ void __launch_bounds__(256) k_lstm(
    const float* __restrict__ obsr,      // [B,T,8] or null
    const _Float16* __restrict__ xp,     // [B,T,fourH] f16 (bias included) or null
    const float* __restrict__ Wx,        // [fourH,8] (obsr path)
    const float* __restrict__ bias,      // [fourH]  (obsr path)
    const _Float16* __restrict__ Whh,    // [fourH,H] f16
    _Float16* __restrict__ hout,         // [B,T,hstride] f16 (pre-offset col) or null
    float* __restrict__ hfinal,          // [B,H] f32 (last step) or null
    int H, int fourH, int reverse, int hstride)
{
  extern __shared__ char smem[];
  const int tid = threadIdx.x, lane = tid & 31, wave = tid >> 5;
  const int b0 = blockIdx.x * 16;
  _Float16* sWh = (_Float16*)smem;                                   // fourH*H*2
  float* sWx = (float*)(smem + (size_t)fourH*H*2);                   // fourH*8*4
  float* sb  = (float*)(smem + (size_t)fourH*H*2 + fourH*32);        // fourH*4
  float* sz  = (float*)(smem + (size_t)fourH*H*2 + fourH*36);        // 16*fourH*4
  _Float16* sh = (_Float16*)(smem + (size_t)fourH*H*2 + fourH*36 + fourH*64); // 16*H*2
  _Float16* sxp0 = (_Float16*)(smem + (size_t)fourH*H*2 + fourH*36 + fourH*64 + 32*H); // xp dbl-buf
  _Float16* sxp1 = sxp0 + 16*fourH;

  if (wave == 0) {
    // stage recurrent weights via Tensor Data Mover
    tdm_load_2d(lds_off(sWh), Whh, (unsigned)H, (unsigned)fourH,
                (unsigned long long)H, TDM_F16);
    __builtin_amdgcn_s_wait_tensorcnt(0);
    if (!obsr) {
      const int t0 = reverse ? (TLEN-1) : 0;
      tdm_load_2d(lds_off(sxp0), xp + ((size_t)b0*TLEN + t0)*fourH, (unsigned)fourH, 16u,
                  (unsigned long long)TLEN*(unsigned long long)fourH, TDM_F16);
    }
  }
  if (obsr) {
    for (int i = tid; i < fourH*8; i += 256) sWx[i] = Wx[i];
    for (int i = tid; i < fourH; i += 256) sb[i] = bias[i];
  }
  for (int i = tid; i < 16*H; i += 256) sh[i] = (_Float16)0.f;
  __syncthreads();

  float creg[8];
#pragma unroll
  for (int p = 0; p < 8; ++p) creg[p] = 0.f;
  const int PJ = H >> 4;                 // gate cols per thread (8 or 4)
  const int r  = tid & 15;
  const int j0 = (tid >> 4) * PJ;
  const int nt_per_wave = fourH >> 7;    // N-tiles per wave (4 or 2)
  const int ksteps = H >> 5;             // K=32 steps (4 or 2)

  for (int step = 0; step < TLEN; ++step) {
    const int t = reverse ? (TLEN - 1 - step) : step;
    if (obsr) {
      for (int i = tid; i < 16*fourH; i += 256) {
        int rr = i / fourH, n = i - rr*fourH;
        const float* ob = obsr + ((size_t)(b0+rr)*TLEN + t)*8;
        float a = sb[n];
#pragma unroll
        for (int k = 0; k < 8; ++k) a += ob[k]*sWx[n*8+k];
        sz[i] = a;
      }
      if (step + 1 < TLEN && tid < 16) {
        const int tn = reverse ? (TLEN-2-step) : (step+1);
        __builtin_prefetch(obsr + ((size_t)(b0+tid)*TLEN + tn)*8, 0, 0);
      }
    } else {
      _Float16* cur = (step & 1) ? sxp1 : sxp0;
      _Float16* nxt = (step & 1) ? sxp0 : sxp1;
      if (wave == 0) {
        __builtin_amdgcn_s_wait_tensorcnt(0);      // current tile landed in LDS
        if (step + 1 < TLEN) {                     // prefetch next timestep's tile
          const int tn = reverse ? (TLEN-2-step) : (step+1);
          tdm_load_2d(lds_off(nxt), xp + ((size_t)b0*TLEN + tn)*fourH, (unsigned)fourH, 16u,
                      (unsigned long long)TLEN*(unsigned long long)fourH, TDM_F16);
        }
      }
      __syncthreads();
      for (int i = tid; i < 16*fourH; i += 256) sz[i] = (float)cur[i];
    }
    __syncthreads();
    for (int w = 0; w < nt_per_wave; ++w) {
      const int nb = (wave*nt_per_wave + w)*16;
      v8f acc = {};
      for (int ks = 0; ks < ksteps; ++ks) {
        v16h a  = load_frag(sh, H, (size_t)(lane & 15), ks*32, lane);
        v16h bb = load_frag(sWh, H, (size_t)(nb + (lane & 15)), ks*32, lane);
        acc = wmma16(a, bb, acc);
      }
      const int col = nb + (lane & 15);
      const int rb = (lane & 16) ? 8 : 0;
#pragma unroll
      for (int v = 0; v < 8; ++v) sz[(rb+v)*fourH + col] += acc[v];
    }
    __syncthreads();
    for (int p = 0; p < PJ; ++p) {
      const int j = j0 + p;
      float zi = sz[r*fourH + j];
      float zf = sz[r*fourH + H + j];
      float zg = sz[r*fourH + 2*H + j];
      float zo = sz[r*fourH + 3*H + j];
      float c = sigmf_(zf)*creg[p] + sigmf_(zi)*tanhf(zg);
      creg[p] = c;
      float h = sigmf_(zo)*tanhf(c);
      sh[r*H + j] = (_Float16)h;
      if (hout) hout[((size_t)(b0+r)*TLEN + t)*hstride + j] = (_Float16)h;
      if (hfinal && step == TLEN-1) hfinal[(size_t)(b0+r)*H + j] = h;
    }
    __syncthreads();
  }
}

// ---------------- layer-2 input projection: [BT,512] = h1[BT,256] @ Wx^T + b ----------------
__global__ void __launch_bounds__(256) k_xproj2(
    const _Float16* __restrict__ A, const _Float16* __restrict__ W,
    const float* __restrict__ bias, _Float16* __restrict__ out)
{
  __shared__ _Float16 sB[64*256];
  const int tid = threadIdx.x, lane = tid & 31, wave = tid >> 5;
  const int nb0 = blockIdx.y * 64;
  if (wave == 0) {                       // TDM-stage the 64x256 B strip
    tdm_load_2d(lds_off(sB), W + (size_t)nb0*256, 256u, 64u, 256ull, TDM_F16);
    __builtin_amdgcn_s_wait_tensorcnt(0);
  }
  __syncthreads();
  const size_t m0 = (size_t)blockIdx.x*128 + (size_t)wave*16;
  v8f acc[4] = {{},{},{},{}};
  for (int ks = 0; ks < 8; ++ks) {
    v16h a = load_frag(A, 256, m0 + (lane & 15), ks*32, lane);
#pragma unroll
    for (int w = 0; w < 4; ++w) {
      v16h b = load_frag(sB, 256, (size_t)(w*16 + (lane & 15)), ks*32, lane);
      acc[w] = wmma16(a, b, acc[w]);
    }
  }
  const int rb = (lane & 16) ? 8 : 0;
#pragma unroll
  for (int w = 0; w < 4; ++w) {
    const int col = nb0 + w*16 + (lane & 15);
    const float bv = bias[col];
#pragma unroll
    for (int v = 0; v < 8; ++v)
      out[(m0 + rb + v)*512 + col] = (_Float16)(acc[w][v] + bv);
  }
}

// ---------------- SVGP precompute: per-GP Cholesky, KL, A = K^-1 - K^-1 S K^-1, alpha ----------------
__global__ void k_gp_prep(
    const float* __restrict__ gZ, const float* __restrict__ gls,
    const float* __restrict__ gos, const float* __restrict__ gum,
    const float* __restrict__ gLraw,
    _Float16* __restrict__ gpB, float* __restrict__ kl_part)
{
  extern __shared__ char smem[];
  float* K    = (float*)smem;
  float* S    = K + 4096;
  float* KiS  = S + 4096;
  float* Kinv = KiS + 4096;
  float* Aw   = Kinv + 4096;
  float* alpha= Aw + 4096;
  const int tid = threadIdx.x;      // 64 threads
  const int d = blockIdx.x;
  const float* Z  = gZ + (size_t)d*64*8;
  const float* ls = gls + d*8;
  const float  os = gos[d];
  const float* m  = gum + (size_t)d*64;
  const float* Lr = gLraw + (size_t)d*64*64;

  for (int i = tid; i < 4096; i += 64) {
    int a = i >> 6, b = i & 63;
    float s = 0.f;
#pragma unroll
    for (int k = 0; k < 8; ++k) { float df = (Z[a*8+k]-Z[b*8+k])/ls[k]; s += df*df; }
    float v = os * __expf(-0.5f*s);
    K[i] = (a==b) ? v + 1e-4f : v;
  }
  for (int i = tid; i < 4096; i += 64) {
    int a = i >> 6, b = i & 63;
    int kk = (a < b) ? a : b;
    float s = 0.f;
    for (int k = 0; k <= kk; ++k) s += Lr[a*64+k]*Lr[b*64+k];
    S[i] = (a==b) ? s + 1e-4f : s;
  }
  __syncthreads();
  for (int j = 0; j < 64; ++j) {
    if (tid == 0) K[j*64+j] = sqrtf(K[j*64+j]);
    __syncthreads();
    if (tid > j) K[tid*64+j] /= K[j*64+j];
    __syncthreads();
    if (tid > j) {
      float lij = K[tid*64+j];
      for (int k = j+1; k <= tid; ++k) K[tid*64+k] -= lij * K[k*64+j];
    }
    __syncthreads();
  }
  for (int i = tid; i < 4096; i += 64) Kinv[i] = ((i>>6)==(i&63)) ? 1.f : 0.f;
  __syncthreads();
  for (int i = 0; i < 64; ++i) {
    float s = Kinv[i*64+tid];
    for (int k = 0; k < i; ++k) s -= K[i*64+k]*Kinv[k*64+tid];
    Kinv[i*64+tid] = s / K[i*64+i];
  }
  for (int i = 63; i >= 0; --i) {
    float s = Kinv[i*64+tid];
    for (int k = i+1; k < 64; ++k) s -= K[k*64+i]*Kinv[k*64+tid];
    Kinv[i*64+tid] = s / K[i*64+i];
  }
  __syncthreads();
  for (int j = 0; j < 64; ++j) {
    float s = 0.f;
    for (int k = 0; k < 64; ++k) s += Kinv[tid*64+k]*S[k*64+j];
    KiS[tid*64+j] = s;
  }
  {
    float s = 0.f;
    for (int k = 0; k < 64; ++k) s += Kinv[tid*64+k]*m[k];
    alpha[tid] = s;
  }
  __syncthreads();
  for (int i = tid; i < 4096; i += 64) Aw[i] = S[i];
  __syncthreads();
  for (int j = 0; j < 64; ++j) {
    if (tid == 0) Aw[j*64+j] = sqrtf(Aw[j*64+j]);
    __syncthreads();
    if (tid > j) Aw[tid*64+j] /= Aw[j*64+j];
    __syncthreads();
    if (tid > j) {
      float lij = Aw[tid*64+j];
      for (int k = j+1; k <= tid; ++k) Aw[tid*64+k] -= lij * Aw[k*64+j];
    }
    __syncthreads();
  }
  for (int j = 0; j < 64; ++j) {
    float s = 0.f;
    for (int k = 0; k < 64; ++k) s += KiS[tid*64+k]*Kinv[k*64+j];
    S[tid*64+j] = Kinv[tid*64+j] - s;
  }
  __syncthreads();
  if (tid == 0) {
    float tr = 0.f, mkm = 0.f, ldK = 0.f, ldS = 0.f;
    for (int i = 0; i < 64; ++i) {
      tr  += KiS[i*64+i];
      mkm += m[i]*alpha[i];
      ldK += __logf(K[i*64+i]);
      ldS += __logf(Aw[i*64+i]);
    }
    kl_part[d] = 0.5f*(tr + mkm - 64.f + 2.f*ldK - 2.f*ldS);
  }
  __syncthreads();
  for (int i = tid; i < 80*64; i += 64) {
    int n = i >> 6, k = i & 63;
    float v = (n < 64) ? S[n*64+k] : ((n == 64) ? alpha[k] : 0.f);
    gpB[(size_t)d*80*64 + i] = (_Float16)v;
  }
}

// ---------------- recognition heads + KL(q(x0)||N(0,I)) + x0 sample ----------------
__global__ void __launch_bounds__(256) k_rec_heads(
    const float* __restrict__ hrec, const float* __restrict__ Wm, const float* __restrict__ bm,
    const float* __restrict__ Wv, const float* __restrict__ bv,
    const float* __restrict__ eps0, float* __restrict__ x0, float* __restrict__ qm0_part)
{
  __shared__ float sred[256];
  const int tid = threadIdx.x;
  const int idx = blockIdx.x*256 + tid;        // B*D = 2048 total
  const int b = idx >> 3, q = idx & 7;
  float m0 = bm[q], vr = bv[q];
  const float* h = hrec + (size_t)b*64;
  for (int k = 0; k < 64; ++k) {
    m0 += h[k]*Wm[q*64+k];
    vr += h[k]*Wv[q*64+k];
  }
  float v0 = softplusf_(vr) + 1e-6f;
  x0[idx] = m0 + sqrtf(v0)*eps0[idx];
  float part = 0.5f*(v0 + m0*m0 - 1.f - __logf(v0));
  float s = block_reduce_sum(part, sred);
  if (tid == 0) qm0_part[blockIdx.x] = s;
}

// ---------------- persistent sequential scan: PostNet + heads + SVGP predict ----------------
// h2 tile TDM'd straight into the padded zcat A-matrix (row = 128 DW, pad 16 DW
// realigns each row at col 8 of the 288-half stride), double-buffered one step ahead.
__global__ void __launch_bounds__(256) k_scan(
    const float* __restrict__ obsr, const float* __restrict__ eps_seq,
    const _Float16* __restrict__ h2, const float* __restrict__ x0,
    const _Float16* __restrict__ W1p, const _Float16* __restrict__ Wmv,
    const _Float16* __restrict__ gpB,
    const float* __restrict__ b1, const float* __restrict__ bm, const float* __restrict__ bv,
    const float* __restrict__ gZ, const float* __restrict__ gls, const float* __restrict__ gos,
    float* __restrict__ scan_part)
{
  extern __shared__ char smem[];
  char* sp = smem;
  _Float16* sW1  = (_Float16*)sp; sp += 147456;   // [128][288]
  _Float16* sWmv = (_Float16*)sp; sp += 4096;     // [16][128]
  _Float16* szc  = (_Float16*)sp; sp += 9216;     // [16][288] buf0
  _Float16* szc2 = (_Float16*)sp; sp += 9216;     // [16][288] buf1
  _Float16* shm  = (_Float16*)sp; sp += 4096;     // [16][128]
  float* shead   = (float*)sp;    sp += 1024;     // [16][16]
  _Float16* skxz = (_Float16*)sp; sp += 16384;    // [8][16][64]
  float* sgp     = (float*)sp;    sp += 40960;    // [8][16][80]
  float* sx      = (float*)sp;    sp += 512;      // [16][8]
  float* sxn     = (float*)sp;    sp += 512;
  float* sb1     = (float*)sp;    sp += 512;
  float* sbm     = (float*)sp;    sp += 256;
  float* sbv     = (float*)sp;    sp += 256;
  float* sZ      = (float*)sp;    sp += 16384;    // [8][64][8]
  float* sls     = (float*)sp;    sp += 256;      // [8][8]
  float* sos     = (float*)sp;    sp += 256;      // [8]
  float* sred    = (float*)sp;    sp += 1024;

  const int tid = threadIdx.x, lane = tid & 31, wave = tid >> 5;
  const int b0 = blockIdx.x * 16;

  for (int i = tid; i < 128*288; i += 256) sW1[i] = W1p[i];
  for (int i = tid; i < 16*128; i += 256) sWmv[i] = Wmv[i];
  for (int i = tid; i < 8*64*8; i += 256) sZ[i] = gZ[i];
  for (int i = tid; i < 16*288; i += 256) { szc[i] = (_Float16)0.f; szc2[i] = (_Float16)0.f; }
  if (tid < 128) sb1[tid] = b1[tid];
  if (tid < 64) sls[tid] = gls[tid];
  if (tid < 8) { sbm[tid] = bm[tid]; sbv[tid] = bv[tid]; sos[tid] = gos[tid]; }
  if (tid < 128) sx[tid] = x0[(size_t)b0*8 + tid];
  __syncthreads();
  if (wave == 0) {    // prime the pipeline: h2 tile for t=0 into zcat buf0, cols 8..263
    tdm_load_2d(lds_off(szc) + 16u, h2 + (size_t)b0*TLEN*256, 256u, 16u,
                (unsigned long long)TLEN*256ull, TDM_F16_PAD288);
  }

  float s_logv = 0.f, s_d2 = 0.f, s_v = 0.f, s_xg = 0.f, s_var = 0.f;

  for (int t = 0; t < TLEN; ++t) {
    _Float16* cz = (t & 1) ? szc2 : szc;
    _Float16* nz = (t & 1) ? szc : szc2;
    if (wave == 0) {
      __builtin_amdgcn_s_wait_tensorcnt(0);        // t's tile landed
      if (t + 1 < TLEN)
        tdm_load_2d(lds_off(nz) + 16u, h2 + ((size_t)b0*TLEN + (t+1))*256, 256u, 16u,
                    (unsigned long long)TLEN*256ull, TDM_F16_PAD288);
    }
    // x_prev columns of the zcat A-matrix + RBF features k(x_prev, Z)
    if (tid < 128) cz[(tid >> 3)*288 + (tid & 7)] = (_Float16)sx[tid];
    for (int i = tid; i < 8192; i += 256) {
      int dd = i >> 10, rr = (i >> 6) & 15, mm = i & 63;
      const float* zr = sZ + (dd*64+mm)*8;
      const float* lr = sls + dd*8;
      const float* xr = sx + rr*8;
      float s = 0.f;
#pragma unroll
      for (int k = 0; k < 8; ++k) { float df = (xr[k]-zr[k])/lr[k]; s += df*df; }
      skxz[i] = (_Float16)(sos[dd]*__expf(-0.5f*s));
    }
    if (t + 1 < TLEN && tid < 16)
      __builtin_prefetch(obsr + ((size_t)(b0+tid)*TLEN + (t+1))*8, 0, 0);
    __syncthreads();
    // B: PostNet hidden = tanh(zcat @ W1^T + b1); each wave = one 16-col N-tile
    {
      const int col = wave*16 + (lane & 15);
      v8f acc;
#pragma unroll
      for (int v = 0; v < 8; ++v) acc[v] = sb1[col];
      for (int ks = 0; ks < 9; ++ks) {
        v16h a = load_frag(cz, 288, (size_t)(lane & 15), ks*32, lane);
        v16h b = load_frag(sW1, 288, (size_t)col, ks*32, lane);
        acc = wmma16(a, b, acc);
      }
      const int rb = (lane & 16) ? 8 : 0;
#pragma unroll
      for (int v = 0; v < 8; ++v) shm[(rb+v)*128 + col] = (_Float16)tanhf(acc[v]);
    }
    __syncthreads();
    // C: heads (wave 0; Wm rows 0-7, Wv rows 8-15 packed) + per-wave SVGP WMMA
    if (wave == 0) {
      const int col = lane & 15;
      float bvv = (col < 8) ? sbm[col] : sbv[col-8];
      v8f acc;
#pragma unroll
      for (int v = 0; v < 8; ++v) acc[v] = bvv;
      for (int ks = 0; ks < 4; ++ks) {
        v16h a = load_frag(shm, 128, (size_t)(lane & 15), ks*32, lane);
        v16h b = load_frag(sWmv, 128, (size_t)col, ks*32, lane);
        acc = wmma16(a, b, acc);
      }
      const int rb = (lane & 16) ? 8 : 0;
#pragma unroll
      for (int v = 0; v < 8; ++v) shead[(rb+v)*16 + col] = acc[v];
    }
    {
      const int dd = wave;  // one GP per wave
      const _Float16* Bg = gpB + (size_t)dd*80*64;
      const _Float16* Ag = skxz + dd*16*64;
      for (int nt = 0; nt < 5; ++nt) {
        const int nb = nt*16;
        v8f acc = {};
        for (int ks = 0; ks < 2; ++ks) {
          v16h a = load_frag(Ag, 64, (size_t)(lane & 15), ks*32, lane);
          v16h b = load_frag(Bg, 64, (size_t)(nb + (lane & 15)), ks*32, lane);
          acc = wmma16(a, b, acc);
        }
        const int col = nb + (lane & 15);
        const int rb = (lane & 16) ? 8 : 0;
#pragma unroll
        for (int v = 0; v < 8; ++v) sgp[(dd*16 + rb+v)*80 + col] = acc[v];
      }
    }
    __syncthreads();
    // D1: sample x_t, entropy + data-fit partials
    if (tid < 128) {
      const int rr = tid >> 3, q = tid & 7;
      float mt = shead[rr*16 + q];
      float vraw = shead[rr*16 + 8 + q];
      float vt = softplusf_(vraw) + 1e-6f;
      float e = eps_seq[((size_t)(b0+rr)*TLEN + t)*8 + q];
      float xt = mt + sqrtf(vt)*e;
      sxn[rr*8+q] = xt;
      s_logv += __logf(vt);
      float y = obsr[((size_t)(b0+rr)*TLEN + t)*8 + q];
      float dfv = y - mt;
      s_d2 += dfv*dfv;
      s_v += vt;
    }
    __syncthreads();
    // D2: GP mean/var partials (mean is column 64; var = os - k.A.k)
    if (tid < 128) {
      const int dd = tid >> 4, rr = tid & 15;
      const float* tr = sgp + (dd*16+rr)*80;
      const _Float16* kr = skxz + (dd*16+rr)*64;
      float dot = 0.f;
      for (int mm2 = 0; mm2 < 64; ++mm2) dot += (float)kr[mm2]*tr[mm2];
      float var = fmaxf(sos[dd] - dot, 1e-8f);
      s_var += var;
      float dx = sxn[rr*8+dd] - tr[64];
      s_xg += dx*dx;
    }
    __syncthreads();
    if (tid < 128) sx[tid] = sxn[tid];
    __syncthreads();
  }
  float r0 = block_reduce_sum(s_logv, sred);
  float r1 = block_reduce_sum(s_d2, sred);
  float r2 = block_reduce_sum(s_v, sred);
  float r3 = block_reduce_sum(s_xg, sred);
  float r4 = block_reduce_sum(s_var, sred);
  if (tid == 0) {
    float* p = scan_part + blockIdx.x*5;
    p[0]=r0; p[1]=r1; p[2]=r2; p[3]=r3; p[4]=r4;
  }
}

// ---------------- deterministic finalize ----------------
__global__ void k_final(const float* __restrict__ qm0_part, const float* __restrict__ kl_part,
                        const float* __restrict__ scan_part, const float* __restrict__ cov_g,
                        float* __restrict__ out)
{
  if (threadIdx.x != 0 || blockIdx.x != 0) return;
  const float log2pi = 1.8378770664093453f;
  float qsum = 0.f, klsum = 0.f;
  for (int i = 0; i < 8; ++i) { qsum += qm0_part[i]; klsum += kl_part[i]; }
  float Slogv=0.f, Sd2=0.f, Sv=0.f, Sxg=0.f, Svar=0.f;
  for (int i = 0; i < 16; ++i) {
    Slogv += scan_part[i*5+0];
    Sd2   += scan_part[i*5+1];
    Sv    += scan_part[i*5+2];
    Sxg   += scan_part[i*5+3];
    Svar  += scan_part[i*5+4];
  }
  const float Bf = 256.f, Tf = 512.f, Df = 8.f, noise = 0.0025f;
  float qm0 = qsum / Bf;
  float KLd = klsum / Tf;
  float Hx = 0.5f*Df + 0.5f*Df*log2pi + 0.5f*Slogv/(Bf*Tf);
  float cg = cov_g[0];
  float datafit = -0.5f*(Sd2 + Sv)/(Bf*cg) - Tf*0.5f*Df*(log2pi + __logf(cg));
  float gpd = -0.5f*(Sxg + Svar)/(Bf*Tf*noise) - 0.5f*Df*(log2pi + __logf(noise));
  out[0] = -qm0 + Hx - KLd + datafit + gpd;
}

// ---------------- host ----------------
extern "C" void kernel_launch(void* const* d_in, const int* in_sizes, int n_in,
                              void* d_out, int out_size, void* d_ws, size_t ws_size,
                              hipStream_t stream)
{
  (void)in_sizes; (void)out_size; (void)ws_size;
  if (n_in < 34) return;
  const float* obsr    = (const float*)d_in[0];
  const float* cov_g   = (const float*)d_in[1];
  const float* eps0    = (const float*)d_in[2];
  const float* eps_seq = (const float*)d_in[3];
  const float* rec_Wx = (const float*)d_in[4];
  const float* rec_Wh = (const float*)d_in[5];
  const float* rec_b  = (const float*)d_in[6];
  const float* rec_Wm = (const float*)d_in[7];
  const float* rec_bm = (const float*)d_in[8];
  const float* rec_Wv = (const float*)d_in[9];
  const float* rec_bv = (const float*)d_in[10];
  const float* e1f_Wx = (const float*)d_in[11];
  const float* e1f_Wh = (const float*)d_in[12];
  const float* e1f_b  = (const float*)d_in[13];
  const float* e1b_Wx = (const float*)d_in[14];
  const float* e1b_Wh = (const float*)d_in[15];
  const float* e1b_b  = (const float*)d_in[16];
  const float* e2f_Wx = (const float*)d_in[17];
  const float* e2f_Wh = (const float*)d_in[18];
  const float* e2f_b  = (const float*)d_in[19];
  const float* e2b_Wx = (const float*)d_in[20];
  const float* e2b_Wh = (const float*)d_in[21];
  const float* e2b_b  = (const float*)d_in[22];
  const float* post_W1 = (const float*)d_in[23];
  const float* post_b1 = (const float*)d_in[24];
  const float* post_Wm = (const float*)d_in[25];
  const float* post_bm = (const float*)d_in[26];
  const float* post_Wv = (const float*)d_in[27];
  const float* post_bv = (const float*)d_in[28];
  const float* gp_Z    = (const float*)d_in[29];
  const float* gp_ls   = (const float*)d_in[30];
  const float* gp_os   = (const float*)d_in[31];
  const float* gp_um   = (const float*)d_in[32];
  const float* gp_Lraw = (const float*)d_in[33];

  char* ws = (char*)d_ws;
  size_t off = 0;
  auto walloc = [&](size_t bytes) -> char* {
    char* p = ws + off;
    off += (bytes + 255) & ~(size_t)255;
    return p;
  };
  float* qm0_part  = (float*)walloc(8*4);
  float* kl_part   = (float*)walloc(8*4);
  float* scan_part = (float*)walloc(80*4);
  _Float16* recWh = (_Float16*)walloc(256*64*2);
  _Float16* e1fWh = (_Float16*)walloc(512*128*2);
  _Float16* e1bWh = (_Float16*)walloc(512*128*2);
  _Float16* e2fWh = (_Float16*)walloc(512*128*2);
  _Float16* e2bWh = (_Float16*)walloc(512*128*2);
  _Float16* e2fWxh= (_Float16*)walloc(512*256*2);
  _Float16* e2bWxh= (_Float16*)walloc(512*256*2);
  _Float16* W1p   = (_Float16*)walloc(128*288*2);
  _Float16* Wmv   = (_Float16*)walloc(16*128*2);
  _Float16* gpB   = (_Float16*)walloc((size_t)8*80*64*2);
  float* hrec = (float*)walloc((size_t)256*64*4);
  float* x0   = (float*)walloc((size_t)256*8*4);
  _Float16* h1   = (_Float16*)walloc((size_t)256*512*256*2);
  _Float16* h2   = (_Float16*)walloc((size_t)256*512*256*2);
  _Float16* xp2f = (_Float16*)walloc((size_t)256*512*512*2);
  _Float16* xp2b = (_Float16*)walloc((size_t)256*512*512*2);

  const int smem_lstm_big = 131072 + 16384 + 2048 + 32768 + 4096;  // 186368 (obsr path)
  const int smem_lstm_e2  = smem_lstm_big + 32768;                 // + xp double-buffer
  const int smem_lstm_rec = 32768 + 8192 + 1024 + 16384 + 2048;    // 60416
  const int smem_gp   = 5*16384 + 256;                             // 82176
  const int smem_scan = 243200 + 9216;                             // + second zcat buffer
  (void)hipFuncSetAttribute((const void*)k_lstm,    hipFuncAttributeMaxDynamicSharedMemorySize, smem_lstm_e2);
  (void)hipFuncSetAttribute((const void*)k_gp_prep, hipFuncAttributeMaxDynamicSharedMemorySize, smem_gp);
  (void)hipFuncSetAttribute((const void*)k_scan,    hipFuncAttributeMaxDynamicSharedMemorySize, smem_scan);

  auto nb256 = [](int n){ return dim3((unsigned)((n + 255) / 256)); };

  k_cast_f16<<<nb256(256*64), 256, 0, stream>>>(rec_Wh, recWh, 256*64);
  k_cast_f16<<<nb256(512*128), 256, 0, stream>>>(e1f_Wh, e1fWh, 512*128);
  k_cast_f16<<<nb256(512*128), 256, 0, stream>>>(e1b_Wh, e1bWh, 512*128);
  k_cast_f16<<<nb256(512*128), 256, 0, stream>>>(e2f_Wh, e2fWh, 512*128);
  k_cast_f16<<<nb256(512*128), 256, 0, stream>>>(e2b_Wh, e2bWh, 512*128);
  k_cast_f16<<<nb256(512*256), 256, 0, stream>>>(e2f_Wx, e2fWxh, 512*256);
  k_cast_f16<<<nb256(512*256), 256, 0, stream>>>(e2b_Wx, e2bWxh, 512*256);
  k_pack_w1<<<nb256(128*288), 256, 0, stream>>>(post_W1, W1p);
  k_pack_wmv<<<nb256(16*128), 256, 0, stream>>>(post_Wm, post_Wv, Wmv);
  k_gp_prep<<<8, 64, smem_gp, stream>>>(gp_Z, gp_ls, gp_os, gp_um, gp_Lraw, gpB, kl_part);

  // recognition LSTM (final h only)
  k_lstm<<<16, 256, smem_lstm_rec, stream>>>(obsr, nullptr, rec_Wx, rec_b, recWh,
                                             nullptr, hrec, 64, 256, 0, 0);
  // encoder layer 1, fwd + bwd -> h1 [B,T,256]
  k_lstm<<<16, 256, smem_lstm_big, stream>>>(obsr, nullptr, e1f_Wx, e1f_b, e1fWh,
                                             h1, nullptr, 128, 512, 0, 256);
  k_lstm<<<16, 256, smem_lstm_big, stream>>>(obsr, nullptr, e1b_Wx, e1b_b, e1bWh,
                                             h1 + 128, nullptr, 128, 512, 1, 256);
  // layer-2 input projections (bulk WMMA GEMMs)
  dim3 gx(1024, 8);
  k_xproj2<<<gx, 256, 0, stream>>>(h1, e2fWxh, e2f_b, xp2f);
  k_xproj2<<<gx, 256, 0, stream>>>(h1, e2bWxh, e2b_b, xp2b);
  // encoder layer 2 (TDM-prefetched xp), fwd + bwd -> h2 [B,T,256]
  k_lstm<<<16, 256, smem_lstm_e2, stream>>>(nullptr, xp2f, nullptr, nullptr, e2fWh,
                                            h2, nullptr, 128, 512, 0, 256);
  k_lstm<<<16, 256, smem_lstm_e2, stream>>>(nullptr, xp2b, nullptr, nullptr, e2bWh,
                                            h2 + 128, nullptr, 128, 512, 1, 256);
  // q(x0) heads + KL + sample
  k_rec_heads<<<8, 256, 0, stream>>>(hrec, rec_Wm, rec_bm, rec_Wv, rec_bv, eps0, x0, qm0_part);
  // sequential scan (TDM-prefetched h2 -> padded zcat)
  k_scan<<<16, 256, smem_scan, stream>>>(obsr, eps_seq, h2, x0, W1p, Wmv, gpB,
                                         post_b1, post_bm, post_bv, gp_Z, gp_ls, gp_os, scan_part);
  k_final<<<1, 32, 0, stream>>>(qm0_part, kl_part, scan_part, cov_g, (float*)d_out);
}
